// LoraAttention_60928406061584
// MI455X (gfx1250) — matrix operand order repaired and checked by reading, not compiled, not run on hardware
//
#include <hip/hip_runtime.h>
#include <stdint.h>

// ---------------- problem constants ----------------
constexpr int kH    = 16;     // heads
constexpr int kD    = 64;     // head dim
constexpr int kC    = 1024;   // model dim
constexpr int kN    = 2048;   // seq len
constexpr int kB    = 2;      // batch
constexpr int kBN   = kB * kN;        // 4096
constexpr int kQKV  = 3 * kH * kD;    // 3072
constexpr int kR    = 10;
constexpr float kLoraScale = 0.1f;    // 1/R
constexpr float kAttScale  = 0.125f;  // D^-0.5

typedef __attribute__((ext_vector_type(16))) __bf16 bf16x16;
typedef __attribute__((ext_vector_type(8)))  float  f32x8;

union FragBF { uint4 q[2]; bf16x16 v; };

__device__ __forceinline__ unsigned short f2bf(float x) {
    unsigned u = __float_as_uint(x);
    unsigned r = 0x7FFFu + ((u >> 16) & 1u);          // round-to-nearest-even
    return (unsigned short)((u + r) >> 16);
}
__device__ __forceinline__ unsigned pack2(float a, float b) {
    return (unsigned)f2bf(a) | ((unsigned)f2bf(b) << 16);
}
__device__ __forceinline__ f32x8 zero8() {
    f32x8 z;
#pragma unroll
    for (int i = 0; i < 8; ++i) z[i] = 0.f;
    return z;
}
__device__ __forceinline__ f32x8 wmma_bf16(const FragBF& a, const FragBF& b, f32x8 c) {
    // D = A(16x32) x B(32x16) + C, f32 accumulate
    return __builtin_amdgcn_wmma_f32_16x16x32_bf16(false, a.v, false, b.v,
                                                   (short)0, c, false, false);
}

// ---------------- kernel 1: f32 -> bf16 convert ----------------
__global__ void k_cvt_bf16(unsigned short* __restrict__ dst,
                           const float* __restrict__ src, int n) {
    int i = blockIdx.x * blockDim.x + threadIdx.x;
    if (i < n) dst[i] = f2bf(src[i]);
}

// ---------------- kernel 2: fold LoRA into QKV weights ----------------
// Wall[r][c] (bf16, 3072x1024), ball[r] (f32)
__global__ void k_build_wall(const float* __restrict__ Wqkv,
                             const float* __restrict__ Wq, const float* __restrict__ bq,
                             const float* __restrict__ Aq, const float* __restrict__ Bq,
                             const float* __restrict__ Wv, const float* __restrict__ bv,
                             const float* __restrict__ Av, const float* __restrict__ Bv,
                             unsigned short* __restrict__ Wall, float* __restrict__ ball) {
    int i = blockIdx.x * blockDim.x + threadIdx.x;   // 3072*1024 threads
    int r = i >> 10, c = i & 1023;
    float w = Wqkv[i];
    if (r < 1024) {
        float s = 0.f;
#pragma unroll
        for (int j = 0; j < kR; ++j) s += Bq[r * kR + j] * Aq[j * kC + c];
        w += Wq[r * kC + c] + kLoraScale * s;
        if (c == 0) ball[r] = bq[r];
    } else if (r < 2048) {
        if (c == 0) ball[r] = 0.f;
    } else {
        int r2 = r - 2048;
        float s = 0.f;
#pragma unroll
        for (int j = 0; j < kR; ++j) s += Bv[r2 * kR + j] * Av[j * kC + c];
        w += Wv[r2 * kC + c] + kLoraScale * s;
        if (c == 0) ball[r] = bv[r2];
    }
    Wall[i] = f2bf(w);
}

// ---------------- kernel 3: QKV GEMM (bf16 WMMA, f32 acc, double-buffered) ----
// qkv = x(4096x1024) @ Wall^T(1024x3072) + ball ; writes Q,K [b][h][n][d], Vt [b][h][d][n]
__global__ void __launch_bounds__(256) k_gemm_qkv(
    const unsigned short* __restrict__ xbf,
    const unsigned short* __restrict__ Wall,
    const float* __restrict__ ball,
    unsigned short* __restrict__ Qo,
    unsigned short* __restrict__ Ko,
    unsigned short* __restrict__ Vto) {
    const int lane = threadIdx.x & 31;
    const int g    = blockIdx.x * 8 + (threadIdx.x >> 5);
    const int m0   = (g & 255) * 16;       // 256 row tiles
    const int n0   = (g >> 8) * 64;        // 48 col groups of 64
    const int lq   = lane & 15, kh = lane >> 4;

    f32x8 acc[4];
#pragma unroll
    for (int t = 0; t < 4; ++t) acc[t] = zero8();

    const unsigned short* arow = xbf + (size_t)(m0 + lq) * kC + kh * 8;
    const unsigned short* wrow[4];
#pragma unroll
    for (int t = 0; t < 4; ++t)
        wrow[t] = Wall + (size_t)(n0 + t * 16 + lq) * kC + kh * 8;

    FragBF fa[2], fb[2][4];
    auto loadk = [&](int kk, int buf) {
        const int K0 = kk * 32;
        fa[buf].q[0] = *(const uint4*)(arow + K0);
        fa[buf].q[1] = *(const uint4*)(arow + K0 + 16);
#pragma unroll
        for (int t = 0; t < 4; ++t) {
            fb[buf][t].q[0] = *(const uint4*)(wrow[t] + K0);
            fb[buf][t].q[1] = *(const uint4*)(wrow[t] + K0 + 16);
        }
    };
    loadk(0, 0);
#pragma unroll 2
    for (int kk = 0; kk < kC / 32; ++kk) {
        const int cur = kk & 1, nxt = cur ^ 1;
        if (kk < kC / 32 - 1) loadk(kk + 1, nxt);   // prefetch next K-step into regs
#pragma unroll
        for (int t = 0; t < 4; ++t) acc[t] = wmma_bf16(fa[cur], fb[cur][t], acc[t]);
    }
    // epilogue: C layout -> lane holds col feat, rows m0 + kh*8 + rr
#pragma unroll
    for (int t = 0; t < 4; ++t) {
        const int feat = n0 + t * 16 + lq;
        const float bias = ball[feat];
        if (feat < 1024) {
            int h = feat >> 6, d = feat & 63;
#pragma unroll
            for (int rr = 0; rr < 8; ++rr) {
                int bn = m0 + kh * 8 + rr, b = bn >> 11, n = bn & 2047;
                Qo[((size_t)(b * kH + h) * kN + n) * kD + d] = f2bf(acc[t][rr] + bias);
            }
        } else if (feat < 2048) {
            int f2 = feat - 1024, h = f2 >> 6, d = f2 & 63;
#pragma unroll
            for (int rr = 0; rr < 8; ++rr) {
                int bn = m0 + kh * 8 + rr, b = bn >> 11, n = bn & 2047;
                Ko[((size_t)(b * kH + h) * kN + n) * kD + d] = f2bf(acc[t][rr] + bias);
            }
        } else {   // V stored transposed: Vt[b][h][d][n] -> contiguous in n => b128 store
            int f2 = feat - 2048, h = f2 >> 6, d = f2 & 63;
            int bn0 = m0 + kh * 8, b = bn0 >> 11, n = bn0 & 2047;
            uint4 pk;
            pk.x = pack2(acc[t][0] + bias, acc[t][1] + bias);
            pk.y = pack2(acc[t][2] + bias, acc[t][3] + bias);
            pk.z = pack2(acc[t][4] + bias, acc[t][5] + bias);
            pk.w = pack2(acc[t][6] + bias, acc[t][7] + bias);
            *(uint4*)&Vto[((size_t)(b * kH + h) * kD + d) * kN + n] = pk;
        }
    }
}

// ---------------- kernel 4: flash attention (transposed orientation) --------
// S^T = K @ Q^T ; online softmax per query column ; O^T = V^T @ P^T
// K-fragments double-buffered across iterations; V loads issued before softmax
// so exp/max VALU work hides their latency.
__global__ void __launch_bounds__(128) k_attn(
    const unsigned short* __restrict__ Qm,
    const unsigned short* __restrict__ Km,
    const unsigned short* __restrict__ Vtm,
    const unsigned char* __restrict__ mask,
    unsigned short* __restrict__ att) {
    __shared__ uint4 Plds[4][16 * 5];   // per-wave 16 q-rows x 40-half padded stride
    const int wid  = threadIdx.x >> 5;
    const int lane = threadIdx.x & 31;
    const int g    = blockIdx.x * 4 + wid;
    const int qt = g & 127, h = (g >> 7) & 15, b = g >> 11;
    const int lq = lane & 15, kh = lane >> 4;
    const int q0 = qt * 16;

    const unsigned short* Qp = Qm + (size_t)(b * kH + h) * kN * kD;
    const unsigned short* Kp = Km + (size_t)(b * kH + h) * kN * kD;
    const unsigned short* Vp = Vtm + (size_t)(b * kH + h) * kD * kN;
    const unsigned long long* mp = (const unsigned long long*)(mask + (size_t)b * kN);

    // Q^T as B-matrix: lane = q column, d = s*32 + kh*8 + {0..7, 16..23}
    FragBF fq[2];
#pragma unroll
    for (int s = 0; s < 2; ++s) {
        const unsigned short* qp = Qp + (size_t)(q0 + lq) * kD + s * 32 + kh * 8;
        fq[s].q[0] = *(const uint4*)qp;
        fq[s].q[1] = *(const uint4*)(qp + 16);
    }

    float m_q = -3.0e38f, l_q = 0.f;
    f32x8 acc[4];
#pragma unroll
    for (int t = 0; t < 4; ++t) acc[t] = zero8();

    FragBF fk[2][2][2];   // [buf][key-subtile][d-step]
    auto loadK = [&](int kb, int buf) {
#pragma unroll
        for (int ks = 0; ks < 2; ++ks) {
            const unsigned short* kp0 = Kp + (size_t)(kb + ks * 16 + lq) * kD + kh * 8;
            fk[buf][ks][0].q[0] = *(const uint4*)kp0;
            fk[buf][ks][0].q[1] = *(const uint4*)(kp0 + 16);
            fk[buf][ks][1].q[0] = *(const uint4*)(kp0 + 32);
            fk[buf][ks][1].q[1] = *(const uint4*)(kp0 + 48);
        }
    };
    loadK(0, 0);

#pragma unroll 2
    for (int kt = 0; kt < kN / 32; ++kt) {
        const int kb = kt * 32;
        const int cur = kt & 1, nxt = cur ^ 1;
        if (kt < kN / 32 - 1) {
            loadK(kb + 32, nxt);                       // next K tile into regs
            __builtin_prefetch(Vp + kb + 32, 0, 0);    // hint next V tile
        }
        // V^T A-fragments for this tile: issue early, consumed after softmax
        FragBF fv[4];
#pragma unroll
        for (int t = 0; t < 4; ++t) {
            const unsigned short* vp = Vp + (size_t)(t * 16 + lq) * kN + kb + kh * 16;
            fv[t].q[0] = *(const uint4*)vp;
            fv[t].q[1] = *(const uint4*)(vp + 8);
        }

        f32x8 st[2];
#pragma unroll
        for (int ks = 0; ks < 2; ++ks) {   // S^T = K @ Q^T
            st[ks] = wmma_bf16(fk[cur][ks][0], fq[0], zero8());
            st[ks] = wmma_bf16(fk[cur][ks][1], fq[1], st[ks]);
        }
        // scale + mask; lane's C rows are keys kb + ks*16 + kh*8 + r
        float p[2][8];
        float mx = -3.0e38f;
#pragma unroll
        for (int ks = 0; ks < 2; ++ks) {
            unsigned long long mb = mp[(kb + ks * 16 + kh * 8) >> 3];
#pragma unroll
            for (int r = 0; r < 8; ++r) {
                float sv = st[ks][r] * kAttScale;
                if (((mb >> (8 * r)) & 0xffull) == 0ull) sv = -3.0e38f;
                p[ks][r] = sv;
                mx = fmaxf(mx, sv);
            }
        }
        mx = fmaxf(mx, __shfl_xor(mx, 16, 32));         // combine lane halves (same q col)
        float mn = fmaxf(m_q, mx);
        float alpha = __expf(m_q - mn);
        m_q = mn;
        float rs = 0.f;
#pragma unroll
        for (int ks = 0; ks < 2; ++ks)
#pragma unroll
            for (int r = 0; r < 8; ++r) { p[ks][r] = __expf(p[ks][r] - mn); rs += p[ks][r]; }
        rs += __shfl_xor(rs, 16, 32);
        l_q = l_q * alpha + rs;
#pragma unroll
        for (int t = 0; t < 4; ++t)
#pragma unroll
            for (int r = 0; r < 8; ++r) acc[t][r] *= alpha;

        // P^T (C layout) -> LDS as P[q][key] (row stride 40 halves, bank-spread)
#pragma unroll
        for (int ks = 0; ks < 2; ++ks) {
            uint4 pk;
            pk.x = pack2(p[ks][0], p[ks][1]);
            pk.y = pack2(p[ks][2], p[ks][3]);
            pk.z = pack2(p[ks][4], p[ks][5]);
            pk.w = pack2(p[ks][6], p[ks][7]);
            Plds[wid][lq * 5 + ks * 2 + kh] = pk;
        }
        // P^T as B-matrix: lane = q column, contiguous 16 keys
        FragBF fp;
        fp.q[0] = Plds[wid][lq * 5 + kh * 2];
        fp.q[1] = Plds[wid][lq * 5 + kh * 2 + 1];

        // O^T += V^T @ P^T
#pragma unroll
        for (int t = 0; t < 4; ++t) acc[t] = wmma_bf16(fv[t], fp, acc[t]);
    }
    // epilogue: O^T lane holds q column; d = t*16 + kh*8 + r contiguous -> b128 store
    const float inv = 1.0f / l_q;
    const int bn = b * kN + q0 + lq;
#pragma unroll
    for (int t = 0; t < 4; ++t) {
        uint4 pk;
        pk.x = pack2(acc[t][0] * inv, acc[t][1] * inv);
        pk.y = pack2(acc[t][2] * inv, acc[t][3] * inv);
        pk.z = pack2(acc[t][4] * inv, acc[t][5] * inv);
        pk.w = pack2(acc[t][6] * inv, acc[t][7] * inv);
        *(uint4*)&att[(size_t)bn * kC + h * kD + t * 16 + kh * 8] = pk;
    }
}

// ---------------- kernel 5: output GEMM (bf16 WMMA, f32 out, double-buffered) --
__global__ void __launch_bounds__(256) k_gemm_out(
    const unsigned short* __restrict__ att,
    const unsigned short* __restrict__ Wob,
    const float* __restrict__ bout,
    float* __restrict__ out) {
    const int lane = threadIdx.x & 31;
    const int g    = blockIdx.x * 8 + (threadIdx.x >> 5);
    const int m0   = (g & 255) * 16;
    const int n0   = (g >> 8) * 64;    // 16 col groups
    const int lq   = lane & 15, kh = lane >> 4;

    f32x8 acc[4];
#pragma unroll
    for (int t = 0; t < 4; ++t) acc[t] = zero8();

    const unsigned short* arow = att + (size_t)(m0 + lq) * kC + kh * 8;
    const unsigned short* wrow[4];
#pragma unroll
    for (int t = 0; t < 4; ++t)
        wrow[t] = Wob + (size_t)(n0 + t * 16 + lq) * kC + kh * 8;

    FragBF fa[2], fb[2][4];
    auto loadk = [&](int kk, int buf) {
        const int K0 = kk * 32;
        fa[buf].q[0] = *(const uint4*)(arow + K0);
        fa[buf].q[1] = *(const uint4*)(arow + K0 + 16);
#pragma unroll
        for (int t = 0; t < 4; ++t) {
            fb[buf][t].q[0] = *(const uint4*)(wrow[t] + K0);
            fb[buf][t].q[1] = *(const uint4*)(wrow[t] + K0 + 16);
        }
    };
    loadk(0, 0);
#pragma unroll 2
    for (int kk = 0; kk < kC / 32; ++kk) {
        const int cur = kk & 1, nxt = cur ^ 1;
        if (kk < kC / 32 - 1) loadk(kk + 1, nxt);
#pragma unroll
        for (int t = 0; t < 4; ++t) acc[t] = wmma_bf16(fa[cur], fb[cur][t], acc[t]);
    }
#pragma unroll
    for (int t = 0; t < 4; ++t) {
        const int c = n0 + t * 16 + lq;
        const float bias = bout[c];
#pragma unroll
        for (int rr = 0; rr < 8; ++rr) {
            int row = m0 + kh * 8 + rr;
            out[(size_t)row * kC + c] = acc[t][rr] + bias;
        }
    }
}

// ---------------- host launch ----------------
extern "C" void kernel_launch(void* const* d_in, const int* in_sizes, int n_in,
                              void* d_out, int out_size, void* d_ws, size_t ws_size,
                              hipStream_t stream) {
    const float* x           = (const float*)d_in[0];
    const unsigned char* msk = (const unsigned char*)d_in[1];
    const float* W_qkv       = (const float*)d_in[2];
    const float* Wq          = (const float*)d_in[3];
    const float* bq          = (const float*)d_in[4];
    const float* Aq          = (const float*)d_in[5];
    const float* Bq          = (const float*)d_in[6];
    const float* Wv          = (const float*)d_in[7];
    const float* bv          = (const float*)d_in[8];
    const float* Av          = (const float*)d_in[9];
    const float* Bv          = (const float*)d_in[10];
    const float* W_out       = (const float*)d_in[11];
    const float* b_out       = (const float*)d_in[12];
    float* out = (float*)d_out;

    size_t off = 0;
    auto alloc = [&](size_t bytes) -> void* {
        void* p = (char*)d_ws + off;
        off += (bytes + 255) & ~(size_t)255;
        return p;
    };
    unsigned short* xbf  = (unsigned short*)alloc((size_t)kBN * kC * 2);
    unsigned short* Wall = (unsigned short*)alloc((size_t)kQKV * kC * 2);
    float*          ball = (float*)alloc((size_t)kQKV * 4);
    unsigned short* Wob  = (unsigned short*)alloc((size_t)kC * kC * 2);
    unsigned short* Qb   = (unsigned short*)alloc((size_t)kB * kH * kN * kD * 2);
    unsigned short* Kb   = (unsigned short*)alloc((size_t)kB * kH * kN * kD * 2);
    unsigned short* Vt   = (unsigned short*)alloc((size_t)kB * kH * kN * kD * 2);
    unsigned short* attb = (unsigned short*)alloc((size_t)kBN * kC * 2);

    k_cvt_bf16<<<(kBN * kC) / 256, 256, 0, stream>>>(xbf, x, kBN * kC);
    k_cvt_bf16<<<(kC * kC) / 256, 256, 0, stream>>>(Wob, W_out, kC * kC);
    k_build_wall<<<(kQKV * kC) / 256, 256, 0, stream>>>(W_qkv, Wq, bq, Aq, Bq,
                                                        Wv, bv, Av, Bv, Wall, ball);
    k_gemm_qkv<<<(256 * 48) / 8, 256, 0, stream>>>(xbf, Wall, ball, Qb, Kb, Vt);
    k_attn<<<(kB * kH * (kN / 16)) / 4, 128, 0, stream>>>(Qb, Kb, Vt, msk, attb);
    k_gemm_out<<<(256 * 16) / 8, 256, 0, stream>>>(attb, Wob, b_out, out);

    (void)in_sizes; (void)n_in; (void)out_size; (void)ws_size;
}
